// gnn_15573551415677
// MI455X (gfx1250) — matrix operand order repaired and checked by reading, not compiled, not run on hardware
//
#include <hip/hip_runtime.h>
#include <hip/hip_fp16.h>

// ---------------------------------------------------------------------------
// GNN block for MI455X (gfx1250, wave32, WMMA).
// Edge MLP (144->144->144->64) over B*N*N=524288 rows = ~53 GFLOP -> run on
// v_wmma_f32_16x16x32_f16. agg reduction fused via register accumulators +
// ds_add_f32 + global_atomic_add_f32. Node MLPs also WMMA.
//
// Round-2: laundered weight pointers (stop LICM -> scratch spills).
// Round-3: launder into address-space(1) so B-fragment loads are
//          global_load_b128 (LOADcnt only) instead of flat_load_b128
//          (which also ties up DScnt / the LDS path); NT hints on the
//          streamed edges input and e output.
// ---------------------------------------------------------------------------

typedef _Float16 v16h __attribute__((ext_vector_type(16)));
typedef _Float16 v8h  __attribute__((ext_vector_type(8)));
typedef float    v8f  __attribute__((ext_vector_type(8)));

typedef const _Float16 __attribute__((address_space(1)))* gwptr;
typedef const v16h     __attribute__((address_space(1)))* gv16p;

#define B_   8
#define N_   256
#define FN_  64
#define FE_  16
#define FO_  64
#define EIN_ 144
#define EINP 160   // K padded to multiple of 32

// Opaque pointer pass-through into the GLOBAL address space: defeats LICM/CSE
// (so weight fragments are re-fetched from L0/L2 instead of being hoisted and
// spilled to scratch) and guarantees global_load (not flat_load) codegen.
__device__ __forceinline__ gwptr launder_g(const _Float16* p)
{
    unsigned long long u = (unsigned long long)p;
    asm volatile("" : "+s"(u));
    return (gwptr)u;
}

// ---------------------------------------------------------------------------
// One MLP layer on a 16-row tile held in per-wave LDS (f16, row-major, kpad).
//   KS  : number of K32 steps (K = 32*KS, zero padded)
//   NT  : number of 16-col output tiles
//   ACT : apply celu (alpha=1)
//   STORE: write activated f16 result back into Xs (for the next layer);
//          otherwise return raw f32 tiles in outT.
// A-fragment layout per CDNA5 ISA (16-bit A 16x32): lanes<16 hold K{0..7,16..23}
// of row M=lane; lanes>=16 hold K{8..15,24..31} of row M=lane-16.
// B fragments are pre-packed in global so each lane's 16 halfs are contiguous.
// ---------------------------------------------------------------------------
template<int KS, int NT, bool ACT, bool STORE>
__device__ __forceinline__ void run_layer(_Float16* Xs, int kpad,
                                          const _Float16* Wp_in,
                                          const float* __restrict__ bias,
                                          int lane, v8f* outT)
{
    gwptr Wp = launder_g(Wp_in);           // global AS, blocks LICM
    const int ml  = lane & 15;
    const int hi8 = (lane & 16) ? 8 : 0;

    v16h afr[KS];
#pragma unroll
    for (int ks = 0; ks < KS; ++ks) {
        const _Float16* p = Xs + ml * kpad + ks * 32 + ((lane & 16) ? 8 : 0);
        v8h lo = *(const v8h*)p;          // ds_load_b128
        v8h hi = *(const v8h*)(p + 16);   // ds_load_b128
        v16h a;
#pragma unroll
        for (int i = 0; i < 8; ++i) { a[i] = lo[i]; a[8 + i] = hi[i]; }
        afr[ks] = a;
    }

#pragma unroll
    for (int c = 0; c < NT; ++c) {
        v8f acc = {};
#pragma unroll
        for (int ks = 0; ks < KS; ++ks) {
            gv16p bp = (gv16p)(Wp + ((size_t)((ks * NT + c) * 32 + lane) << 4));
            v16h bfr = *bp;               // 2x global_load_b128, contiguous
            acc = __builtin_amdgcn_wmma_f32_16x16x32_f16(
                false, afr[ks], false, bfr, (short)0, acc, false, false);
        }
        float bv = bias[c * 16 + ml];
#pragma unroll
        for (int r = 0; r < 8; ++r) {
            float x = acc[r] + bv;
            if (ACT) x = x > 0.f ? x : (__expf(x) - 1.f);   // celu, alpha=1
            if (STORE) Xs[(r + hi8) * kpad + c * 16 + ml] = (_Float16)x;
            else       outT[c][r] = x;
        }
    }
}

// ---------------------------------------------------------------------------
// Pack a fp32 weight W[K][Nout] into fragment-native f16:
//   dst[(((ks*NT + c)*32 + lane)*16 + h] = W[ks*32 + (h<8?h:h+8) + (lane>=16?8:0)][c*16 + (lane&15)]
// (zero for k >= K -> handles K=144 padded to 160)
// ---------------------------------------------------------------------------
__global__ void pack_kernel(const float* __restrict__ W, _Float16* __restrict__ dst,
                            int K, int Nout, int KS, int NT)
{
    int total = KS * NT * 512;
    for (int u = blockIdx.x * blockDim.x + threadIdx.x; u < total;
         u += gridDim.x * blockDim.x) {
        int h = u & 15;
        int l = (u >> 4) & 31;
        int rest = u >> 9;
        int c = rest % NT;
        int ks = rest / NT;
        int n = c * 16 + (l & 15);
        int k = ks * 32 + (h < 8 ? h : h + 8) + ((l & 16) ? 8 : 0);
        float v = (k < K) ? W[k * Nout + n] : 0.f;
        dst[u] = (_Float16)v;
    }
}

// nodes -> f16, and zero agg accumulator (first B*N*FO floats of d_out)
__global__ void prep_kernel(const float* __restrict__ nodes,
                            _Float16* __restrict__ nodes_h,
                            float* __restrict__ agg, int n)
{
    int i = blockIdx.x * blockDim.x + threadIdx.x;
    if (i < n) { nodes_h[i] = (_Float16)nodes[i]; agg[i] = 0.f; }
}

// ---------------------------------------------------------------------------
// Edge MLP + mask + e store + agg reduction.
// grid = 512 blocks: bit[1:0]=i-split(4), bits[5:2]=j-tile(16), bits[8:6]=b(8)
// 256 threads = 8 waves; wave w handles i = sp*64 + q*8 + w, q=0..7, for its
// block's 16-row j tile.
// ---------------------------------------------------------------------------
__global__ __launch_bounds__(256) void edge_mlp_kernel(
    const float* __restrict__ edges, const int* __restrict__ skel,
    const _Float16* __restrict__ nodes_h,
    const _Float16* __restrict__ ew0p, const float* __restrict__ eb0,
    const _Float16* __restrict__ ew1p, const float* __restrict__ eb1,
    const _Float16* __restrict__ ew2p, const float* __restrict__ eb2,
    float* __restrict__ e_out, float* __restrict__ agg_out)
{
    __shared__ _Float16 XsAll[8][16 * EINP];   // 40 KB: per-wave X/Y tiles
    __shared__ float aggS[16 * 64];            // 4 KB: per-block agg partial

    const int tid = threadIdx.x, lane = tid & 31, wv = tid >> 5;
    const int sp = blockIdx.x & 3;
    const int jt = (blockIdx.x >> 2) & 15;
    const int b  = blockIdx.x >> 6;
    const int jbase = jt * 16;
    const int ml = lane & 15, hi8 = (lane & 16) ? 8 : 0;
    _Float16* Xs = XsAll[wv];

    // zero K-pad columns [144,160) once; zero agg partial
    for (int u = lane; u < 256; u += 32)
        Xs[(u >> 4) * EINP + EIN_ + (u & 15)] = (_Float16)0.f;
    for (int u = tid; u < 1024; u += 256) aggS[u] = 0.f;
    __syncthreads();

    v8f agg[4] = {};

#pragma unroll 1
    for (int q = 0; q < 8; ++q) {
        const int i = sp * 64 + q * 8 + wv;

        // build X = [edges(16) | nodes[j](64) | nodes[i](64)] for 16 rows (j)
        const float*    eg = edges + (((size_t)b * N_ + i) * N_ + jbase) * FE_;
        const _Float16* nj = nodes_h + ((size_t)b * N_ + jbase) * FN_;
        const _Float16* ni = nodes_h + ((size_t)b * N_ + i) * FN_;
#pragma unroll 1
        for (int u = lane; u < 16 * EIN_; u += 32) {
            int m = u / EIN_;
            int f = u - m * EIN_;
            _Float16 v;
            if (f < FE_)             v = (_Float16)__builtin_nontemporal_load(&eg[m * FE_ + f]);
            else if (f < FE_ + FN_)  v = nj[m * FN_ + (f - FE_)];
            else                     v = ni[f - FE_ - FN_];
            Xs[m * EINP + f] = v;
        }

        run_layer<5, 9, true,  true >(Xs, EINP, ew0p, eb0, lane, nullptr);
        run_layer<5, 9, true,  true >(Xs, EINP, ew1p, eb1, lane, nullptr);
        v8f y[4];
        run_layer<5, 4, false, false>(Xs, EINP, ew2p, eb2, lane, y);

        // skeleton mask per row j
        float sk[8];
        const int* skp = skel + ((size_t)b * N_ + i) * N_ + jbase;
#pragma unroll
        for (int r = 0; r < 8; ++r) sk[r] = (float)skp[r + hi8];

        float* ep = e_out + (((size_t)b * N_ + i) * N_ + jbase) * FO_;
#pragma unroll
        for (int c = 0; c < 4; ++c) {
#pragma unroll
            for (int r = 0; r < 8; ++r) {
                float v = y[c][r] * sk[r];
                // streamed output, never re-read: non-temporal store
                __builtin_nontemporal_store(v, &ep[(r + hi8) * FO_ + c * 16 + ml]);
                agg[c][r] += v;                          // sum over i in regs
            }
        }
    }

    // 8 waves -> block partial in LDS, then 4 i-split partials -> global atomics
#pragma unroll
    for (int c = 0; c < 4; ++c)
#pragma unroll
        for (int r = 0; r < 8; ++r)
            atomicAdd(&aggS[(r + hi8) * 64 + c * 16 + ml], agg[c][r]);
    __syncthreads();
    for (int u = tid; u < 1024; u += 256)
        atomicAdd(&agg_out[((size_t)b * N_ + jbase + (u >> 6)) * FO_ + (u & 63)],
                  aggS[u]);
}

// ---------------------------------------------------------------------------
// out_nodes = mlp3(nodes; mw) + mlp3(agg; nw).   io = agg in / out_nodes out.
// 16 blocks x 8 waves, one 16-row tile per wave (2048 rows total).
// ---------------------------------------------------------------------------
__global__ __launch_bounds__(256) void node_mlp_kernel(
    const _Float16* __restrict__ nodes_h,
    const _Float16* __restrict__ nw0p, const float* __restrict__ nb0,
    const _Float16* __restrict__ nw1p, const float* __restrict__ nb1,
    const _Float16* __restrict__ nw2p, const float* __restrict__ nb2,
    const _Float16* __restrict__ mw0p, const float* __restrict__ mb0,
    const _Float16* __restrict__ mw1p, const float* __restrict__ mb1,
    const _Float16* __restrict__ mw2p, const float* __restrict__ mb2,
    float* __restrict__ io)
{
    __shared__ _Float16 XsAll[8][16 * 64];   // 16 KB
    const int tid = threadIdx.x, lane = tid & 31, wv = tid >> 5;
    const int ml = lane & 15, hi8 = (lane & 16) ? 8 : 0;
    const int rbase = (blockIdx.x * 8 + wv) * 16;
    _Float16* Xs = XsAll[wv];

    // path 1: nodes -> mw MLP
    for (int u = lane; u < 1024; u += 32)
        Xs[u] = nodes_h[(size_t)rbase * 64 + u];
    run_layer<2, 4, true,  true >(Xs, 64, mw0p, mb0, lane, nullptr);
    run_layer<2, 4, true,  true >(Xs, 64, mw1p, mb1, lane, nullptr);
    v8f om[4];
    run_layer<2, 4, false, false>(Xs, 64, mw2p, mb2, lane, om);

    // path 2: agg -> nw MLP (reads io before writing it; wave owns its rows)
    for (int u = lane; u < 1024; u += 32)
        Xs[u] = (_Float16)io[(size_t)rbase * 64 + u];
    run_layer<2, 4, true,  true >(Xs, 64, nw0p, nb0, lane, nullptr);
    run_layer<2, 4, true,  true >(Xs, 64, nw1p, nb1, lane, nullptr);
    v8f on[4];
    run_layer<2, 4, false, false>(Xs, 64, nw2p, nb2, lane, on);

#pragma unroll
    for (int c = 0; c < 4; ++c)
#pragma unroll
        for (int r = 0; r < 8; ++r)
            io[((size_t)rbase + r + hi8) * 64 + c * 16 + ml] = om[c][r] + on[c][r];
}

// ---------------------------------------------------------------------------
extern "C" void kernel_launch(void* const* d_in, const int* in_sizes, int n_in,
                              void* d_out, int out_size, void* d_ws, size_t ws_size,
                              hipStream_t stream)
{
    (void)in_sizes; (void)n_in; (void)out_size; (void)ws_size;

    const float* nodes = (const float*)d_in[0];
    const float* edges = (const float*)d_in[1];
    const int*   skel  = (const int*)d_in[2];
    const float* ew0 = (const float*)d_in[3];  const float* eb0 = (const float*)d_in[4];
    const float* ew1 = (const float*)d_in[5];  const float* eb1 = (const float*)d_in[6];
    const float* ew2 = (const float*)d_in[7];  const float* eb2 = (const float*)d_in[8];
    const float* nw0 = (const float*)d_in[9];  const float* nb0 = (const float*)d_in[10];
    const float* nw1 = (const float*)d_in[11]; const float* nb1 = (const float*)d_in[12];
    const float* nw2 = (const float*)d_in[13]; const float* nb2 = (const float*)d_in[14];
    const float* mw0 = (const float*)d_in[15]; const float* mb0 = (const float*)d_in[16];
    const float* mw1 = (const float*)d_in[17]; const float* mb1 = (const float*)d_in[18];
    const float* mw2 = (const float*)d_in[19]; const float* mb2 = (const float*)d_in[20];

    float* out   = (float*)d_out;                     // out_nodes / agg accumulator
    float* e_out = out + (size_t)B_ * N_ * FO_;       // e [B,N,N,FO]

    // workspace layout (bytes)
    char* ws = (char*)d_ws;
    _Float16* nodes_h = (_Float16*)(ws + 0);          // 262144 B
    _Float16* ew0p = (_Float16*)(ws + 262144);        //  46080 B (KS=5,NT=9)
    _Float16* ew1p = (_Float16*)(ws + 308224);        //  46080 B
    _Float16* ew2p = (_Float16*)(ws + 354304);        //  20480 B (KS=5,NT=4)
    _Float16* nw0p = (_Float16*)(ws + 374784);        //   8192 B (KS=2,NT=4)
    _Float16* nw1p = (_Float16*)(ws + 382976);
    _Float16* nw2p = (_Float16*)(ws + 391168);
    _Float16* mw0p = (_Float16*)(ws + 399360);
    _Float16* mw1p = (_Float16*)(ws + 407552);
    _Float16* mw2p = (_Float16*)(ws + 415744);        // end 423936 B

    prep_kernel<<<512, 256, 0, stream>>>(nodes, nodes_h, out, B_ * N_ * FO_);

    pack_kernel<<<90, 256, 0, stream>>>(ew0, ew0p, EIN_, EIN_, 5, 9);
    pack_kernel<<<90, 256, 0, stream>>>(ew1, ew1p, EIN_, EIN_, 5, 9);
    pack_kernel<<<40, 256, 0, stream>>>(ew2, ew2p, EIN_, FO_, 5, 4);
    pack_kernel<<<16, 256, 0, stream>>>(nw0, nw0p, FO_, FO_, 2, 4);
    pack_kernel<<<16, 256, 0, stream>>>(nw1, nw1p, FO_, FO_, 2, 4);
    pack_kernel<<<16, 256, 0, stream>>>(nw2, nw2p, FO_, FO_, 2, 4);
    pack_kernel<<<16, 256, 0, stream>>>(mw0, mw0p, FN_, FO_, 2, 4);
    pack_kernel<<<16, 256, 0, stream>>>(mw1, mw1p, FO_, FO_, 2, 4);
    pack_kernel<<<16, 256, 0, stream>>>(mw2, mw2p, FO_, FO_, 2, 4);

    edge_mlp_kernel<<<512, 256, 0, stream>>>(edges, skel, nodes_h,
                                             ew0p, eb0, ew1p, eb1, ew2p, eb2,
                                             e_out, out);

    node_mlp_kernel<<<16, 256, 0, stream>>>(nodes_h,
                                            nw0p, nb0, nw1p, nb1, nw2p, nb2,
                                            mw0p, mb0, mw1p, mb1, mw2p, mb2,
                                            out);
}
